// RIMCell_48318382080198
// MI455X (gfx1250) — compile-verified, matrix operand order checked by reading, not run on hardware
//
#include <hip/hip_runtime.h>
#include <hip/hip_bf16.h>

// ---------------------------------------------------------------------------
// RIM cell forward for MI455X (gfx1250, wave32, WMMA).
// Compute-bound on 4 big per-unit GEMMs (~270 GFLOP total) -> bf16 WMMA
// (v_wmma_f32_16x16x32_bf16, f32 accumulate). Weights repacked once per launch
// into fragment-major bf16 tiles; activations staged in bf16 with K padded to
// multiples of 32 so A fragments are two b128 loads per lane.
// ---------------------------------------------------------------------------

#define DEVI __device__ __forceinline__

typedef __attribute__((ext_vector_type(16))) __bf16        v16bf;
typedef __attribute__((ext_vector_type(8)))  float         v8f;
typedef __attribute__((ext_vector_type(8)))  unsigned int  v8u;

// ---- problem sizes (fixed by the reference) -------------------------------
constexpr int BATCH = 4096;
constexpr int NU    = 6;
constexpr int IN    = 600;
constexpr int HH    = 600;
constexpr int VD    = 400;
constexpr int H4    = 2400;   // 4*H
constexpr int NHC   = 4;      // comm heads
constexpr int KDC   = 32;     // comm head dim

// padded K dims (multiples of 32)
constexpr int KIN_P = 608;    // 600 -> 608 (x, hs, h_b as GEMM-A)
constexpr int KV_P  = 416;    // 400 -> 416 (inputs as GEMM-A)
constexpr int KC    = 2400;   // already multiple of 32 (ctx as GEMM-A)
constexpr int NH_P  = 608;    // Wco N: 600 -> 608

// packed-weight tile counts (16x16 tiles, 256 bf16 = 512B per tile)
constexpr int KT_IN = KIN_P / 16;  // 38
constexpr int KT_V  = KV_P  / 16;  // 26
constexpr int KT_C  = KC    / 16;  // 150

// ---- bf16 helpers ---------------------------------------------------------
DEVI unsigned short f2bf(float f) {
  unsigned int u = __float_as_uint(f);
  unsigned int r = (u + 0x7FFFu + ((u >> 16) & 1u)) >> 16;   // RNE
  return (unsigned short)r;
}
DEVI float bf2f(unsigned short h) {
  return __uint_as_float(((unsigned int)h) << 16);
}

// ---- weight repack: f32 (K,N) row-major -> fragment-major bf16 tiles ------
// Tile t = nt*KT + kt holds K-rows [16kt,16kt+16) x N-cols [16nt,16nt+16).
// In-tile: lane 0-15 -> N=lane, K-sub 0..7 ; lane 16-31 -> N=lane-16, K-sub 8..15
// (matches the v_wmma 16-bit B-operand striping; K-tiles 2s / 2s+1 form the
//  low / high 4 VGPRs of one 32-K fragment).
__global__ __launch_bounds__(256)
void repack_w_k(const float* __restrict__ src, long sbatch,
                unsigned short* __restrict__ dst, long dbatch,
                int K, int N, int KT) {
  long u = blockIdx.y;
  long t = blockIdx.x;
  int  kt = (int)(t % KT);
  int  nt = (int)(t / KT);
  int  e    = threadIdx.x;        // 0..255
  int  lane = e >> 3;
  int  j    = e & 7;
  int  n    = nt * 16 + (lane & 15);
  int  r    = ((lane >> 4) << 3) + j;
  int  k    = kt * 16 + r;
  float v = (k < K && n < N) ? src[u * sbatch + (long)k * N + n] : 0.f;
  dst[u * dbatch + t * 256 + e] = f2bf(v);
}

// ---- activation pack: f32 (rows,C) -> bf16 (rows,Cp) zero-padded ----------
__global__ __launch_bounds__(256)
void pack_act_k(const float* __restrict__ src, unsigned short* __restrict__ dst,
                long rows, int C, int Cp) {
  long i = (long)blockIdx.x * blockDim.x + threadIdx.x;
  if (i >= rows * (long)Cp) return;
  int  col = (int)(i % Cp);
  long r   = i / Cp;
  dst[i] = (col < C) ? f2bf(src[r * (long)C + col]) : (unsigned short)0;
}

// ---- WMMA fragment loads --------------------------------------------------
DEVI v16bf load_a(const unsigned short* __restrict__ A, long lda, long row,
                  int s, int lane) {
  // ISA 16-bit A layout: lanes 0-15 hold K 0..7 / 16..23, lanes 16-31 hold
  // K 8..15 / 24..31 of the 32-K step, all for row M=lane&15.
  const unsigned short* p = A + row * lda + s * 32 + ((lane >> 4) << 3);
  const uint4* q = (const uint4*)p;
  uint4 x = q[0];
  uint4 y = q[2];                  // +16 bf16 = +32 bytes
  v8u u = {x.x, x.y, x.z, x.w, y.x, y.y, y.z, y.w};
  return __builtin_bit_cast(v16bf, u);
}

DEVI v16bf load_b(const unsigned short* __restrict__ Bp, long tile, int lane) {
  const uint4* q0 = (const uint4*)(Bp + (tile * 32 + lane) * 8);
  const uint4* q1 = (const uint4*)(Bp + ((tile + 1) * 32 + lane) * 8);
  uint4 x = *q0;
  uint4 y = *q1;
  v8u u = {x.x, x.y, x.z, x.w, y.x, y.y, y.z, y.w};
  return __builtin_bit_cast(v16bf, u);
}

DEVI v8f wmma_bf16(v16bf a, v16bf b, v8f c) {
  return __builtin_amdgcn_wmma_f32_16x16x32_bf16(
      /*neg_a=*/false, a, /*neg_b=*/false, b,
      /*c_mod=*/(short)0, c, /*reuse_a=*/false, /*reuse_b=*/false);
}

// One K-sweep: wave computes 32Mx32N via 2x2 WMMA tiles.
DEVI void gemm_pass(const unsigned short* __restrict__ A, long lda, int ks,
                    const unsigned short* __restrict__ Bp, int KT,
                    int mbase, int nt0, int lane, v8f acc[4]) {
  long r0 = mbase + (lane & 15);
  for (int s = 0; s < ks; ++s) {
    if (s + 1 < ks) {
      __builtin_prefetch(A + r0 * lda + (s + 1) * 32, 0, 3);
      __builtin_prefetch(Bp + (((long)nt0 * KT + 2 * (s + 1)) * 32 + lane) * 8, 0, 3);
    }
    v16bf a0 = load_a(A, lda, r0,      s, lane);
    v16bf a1 = load_a(A, lda, r0 + 16, s, lane);
    v16bf b0 = load_b(Bp, (long)nt0 * KT + 2 * s,       lane);
    v16bf b1 = load_b(Bp, (long)(nt0 + 1) * KT + 2 * s, lane);
    acc[0] = wmma_bf16(a0, b0, acc[0]);
    acc[1] = wmma_bf16(a0, b1, acc[1]);
    acc[2] = wmma_bf16(a1, b0, acc[2]);
    acc[3] = wmma_bf16(a1, b1, acc[3]);
  }
}

// ---- generic GEMM (optional fused second pass; f32 or bf16 store) ---------
// grid = (Ntiles/2, M/256, units); block = 256 (8 waves, wave tile 32x32).
__global__ __launch_bounds__(256)
void wmma_gemm_k(const unsigned short* __restrict__ A0, long a0b, long lda0, int ks0,
                 const unsigned short* __restrict__ B0, long b0b, int kt0,
                 const unsigned short* __restrict__ A1, long a1b, long lda1, int ks1,
                 const unsigned short* __restrict__ B1, long b1b, int kt1,
                 float* __restrict__ Cf, unsigned short* __restrict__ Cbf,
                 long cb, long ldc, int nmax) {
  int u    = blockIdx.z;
  int lane = threadIdx.x & 31;
  int w    = threadIdx.x >> 5;
  int mbase = blockIdx.y * 256 + w * 32;
  int nt0   = blockIdx.x * 2;

  v8f acc[4];
  for (int i = 0; i < 4; ++i)
    acc[i] = v8f{0.f, 0.f, 0.f, 0.f, 0.f, 0.f, 0.f, 0.f};

  gemm_pass(A0 + (long)u * a0b, lda0, ks0, B0 + (long)u * b0b, kt0,
            mbase, nt0, lane, acc);
  if (A1)
    gemm_pass(A1 + (long)u * a1b, lda1, ks1, B1 + (long)u * b1b, kt1,
              mbase, nt0, lane, acc);

  // C/D layout: VGPR v, lanes 0-15 -> M=v, N=lane; lanes 16-31 -> M=8+v.
  int rofs = (lane >> 4) << 3;
  int cl   = lane & 15;
  for (int t = 0; t < 4; ++t) {
    int msub = mbase + (t >> 1) * 16;
    int col  = (nt0 + (t & 1)) * 16 + cl;
    if (col >= nmax) continue;
    if (Cf) {
      float* d = Cf + (long)u * cb;
#pragma unroll
      for (int v = 0; v < 8; ++v)
        d[(long)(msub + rofs + v) * ldc + col] = acc[t][v];
    } else {
      unsigned short* d = Cbf + (long)u * cb;
#pragma unroll
      for (int v = 0; v < 8; ++v)
        d[(long)(msub + rofs + v) * ldc + col] = f2bf(acc[t][v]);
    }
  }
}

// ---- final GEMM: h_new = ctx@Wco + h_b; hs_out = mask ? h_new : hs --------
__global__ __launch_bounds__(256)
void wmma_gemm_final_k(const unsigned short* __restrict__ ctx,
                       const unsigned short* __restrict__ Wco, long wcob,
                       const unsigned short* __restrict__ hb,
                       const float* __restrict__ hs_in,
                       const float* __restrict__ mask,
                       float* __restrict__ out) {
  int u    = blockIdx.z;
  int lane = threadIdx.x & 31;
  int w    = threadIdx.x >> 5;
  int mbase = blockIdx.y * 256 + w * 32;
  int nt0   = blockIdx.x * 2;

  v8f acc[4];
  for (int i = 0; i < 4; ++i)
    acc[i] = v8f{0.f, 0.f, 0.f, 0.f, 0.f, 0.f, 0.f, 0.f};

  gemm_pass(ctx + (long)u * KC, (long)NU * KC, KC / 32,
            Wco + (long)u * wcob, KT_C, mbase, nt0, lane, acc);

  int rofs = (lane >> 4) << 3;
  int cl   = lane & 15;
  for (int t = 0; t < 4; ++t) {
    int msub = mbase + (t >> 1) * 16;
    int col  = (nt0 + (t & 1)) * 16 + cl;
    if (col >= HH) continue;
#pragma unroll
    for (int v = 0; v < 8; ++v) {
      long r   = msub + rofs + v;
      long idx = r * NU + u;
      float hbv = bf2f(hb[idx * (long)KIN_P + col]);
      bool  mk  = mask[idx] > 0.5f;
      out[idx * (long)HH + col] =
          mk ? (acc[t][v] + hbv) : hs_in[idx * (long)HH + col];
    }
  }
}

// ---- input attention: scores -> top-4 mask -> inputs (bf16, K-padded) -----
// Exact forward: key row 1 / val row 1 are zero, so probs[...,0]=sigmoid(s)
// and inputs[b,u,:] = mask * sigmoid(s) * val0.
__global__ __launch_bounds__(32)
void scores_inputs_k(const float* __restrict__ qry, const float* __restrict__ key,
                     const float* __restrict__ val, float* __restrict__ mask,
                     unsigned short* __restrict__ inputs) {
  int b    = blockIdx.x;
  int lane = threadIdx.x;
  float s = 0.f;
  if (lane < NU) {
    const float* q = qry + ((long)b * NU + lane) * 64;
    const float* k = key + (long)b * 64;
    float a = 0.f;
    for (int j = 0; j < 64; ++j) a += q[j] * k[j];
    s = a * 0.125f;                       // 1/sqrt(64)
  }
  float sc[NU];
  for (int m = 0; m < NU; ++m) sc[m] = __shfl(s, m, 32);

  float wgt[NU];
  for (int m = 0; m < NU; ++m) {
    int rank = 0;                         // jax top_k: ties -> first index
    for (int j = 0; j < NU; ++j)
      if (sc[j] > sc[m] || (sc[j] == sc[m] && j < m)) rank++;
    float mk = (rank < 4) ? 1.f : 0.f;
    float p0 = 1.f / (1.f + __expf(-sc[m]));
    wgt[m] = mk * p0;
    if (lane == 0) mask[(long)b * NU + m] = mk;
  }
  for (int c = lane; c < KV_P; c += 32) {
    float vv = (c < VD) ? val[(long)b * VD + c] : 0.f;
    for (int m = 0; m < NU; ++m)
      inputs[((long)b * NU + m) * KV_P + c] = f2bf(wgt[m] * vv);
  }
}

// ---- LSTM gates; writes cs_out to d_out and h_b (bf16, K-padded) ----------
__global__ __launch_bounds__(128)
void lstm_k(const unsigned short* __restrict__ preact,
            const float* __restrict__ cs, const float* __restrict__ mask,
            unsigned short* __restrict__ hb, float* __restrict__ cs_out) {
  long row = blockIdx.x;                  // b*NU + u
  float mk = mask[row];
  const unsigned short* p = preact + row * (long)H4;
  for (int c = threadIdx.x; c < KIN_P; c += blockDim.x) {
    if (c < HH) {
      float pi = bf2f(p[c]);
      float pf = bf2f(p[HH + c]);
      float po = bf2f(p[2 * HH + c]);
      float pg = bf2f(p[3 * HH + c]);
      float it = 1.f / (1.f + __expf(-pi));
      float ft = 1.f / (1.f + __expf(-pf));
      float ot = 1.f / (1.f + __expf(-po));
      float gt = tanhf(pg);
      float c0 = cs[row * (long)HH + c];
      float ct = c0 * ft + it * gt;
      float ht = ot * tanhf(ct);
      hb[row * (long)KIN_P + c]     = f2bf(ht);
      cs_out[row * (long)HH + c]    = (mk > 0.5f) ? ct : c0;
    } else {
      hb[row * (long)KIN_P + c] = 0;      // K-pad
    }
  }
}

// ---- 4-head 6x6 communication attention -----------------------------------
__global__ __launch_bounds__(64)
void cattn_k(const float* __restrict__ ck, const float* __restrict__ cq,
             const unsigned short* __restrict__ cv, const float* __restrict__ mask,
             unsigned short* __restrict__ ctx) {
  __shared__ float p[NU][NU];
  __shared__ float mrow[NU];
  int bh = blockIdx.x;
  int b  = bh >> 2;
  int h  = bh & 3;
  int t  = threadIdx.x;
  if (t < NU * NU) {
    int uu = t / NU, m = t % NU;
    const float* q  = cq + ((long)b * NU + uu) * (NHC * KDC) + h * KDC;
    const float* kk = ck + ((long)b * NU + m)  * (NHC * KDC) + h * KDC;
    float a = 0.f;
    for (int k = 0; k < KDC; ++k) a += q[k] * kk[k];
    p[uu][m] = a * 0.17677669529f;        // 1/sqrt(32)
  }
  if (t < NU) mrow[t] = mask[(long)b * NU + t];
  __syncthreads();
  if (t < NU) {
    float mx = p[t][0];
    for (int m = 1; m < NU; ++m) mx = fmaxf(mx, p[t][m]);
    float e[NU], sum = 0.f;
    for (int m = 0; m < NU; ++m) { e[m] = __expf(p[t][m] - mx); sum += e[m]; }
    float inv = mrow[t] / sum;
    for (int m = 0; m < NU; ++m) p[t][m] = e[m] * inv;
  }
  __syncthreads();
  for (int uu = 0; uu < NU; ++uu) {
    float pm[NU];
    for (int m = 0; m < NU; ++m) pm[m] = p[uu][m];
    for (int d = t; d < HH; d += 64) {
      float a = 0.f;
      for (int m = 0; m < NU; ++m)
        a += pm[m] * bf2f(cv[((long)b * NU + m) * (long)H4 + h * HH + d]);
      ctx[((long)b * NU + uu) * (long)H4 + h * HH + d] = f2bf(a);
    }
  }
}

// ===========================================================================
extern "C" void kernel_launch(void* const* d_in, const int* in_sizes, int n_in,
                              void* d_out, int out_size, void* d_ws, size_t ws_size,
                              hipStream_t stream) {
  const float* x    = (const float*)d_in[0];
  const float* hs   = (const float*)d_in[1];
  const float* cs   = (const float*)d_in[2];
  const float* Wk   = (const float*)d_in[3];
  const float* Wv   = (const float*)d_in[4];
  const float* Wq   = (const float*)d_in[5];
  const float* Wi2h = (const float*)d_in[6];
  const float* Wh2h = (const float*)d_in[7];
  const float* Wck  = (const float*)d_in[8];
  const float* Wcq  = (const float*)d_in[9];
  const float* Wcv  = (const float*)d_in[10];
  const float* Wco  = (const float*)d_in[11];
  float* out = (float*)d_out;

  // ---- workspace carve-up (all 256B aligned) ----
  char*  base = (char*)d_ws;
  size_t off  = 0;
  auto take = [&](size_t bytes) -> void* {
    void* p = base + off;
    off = (off + bytes + 255) & ~(size_t)255;
    return p;
  };
  const long TWk   = (long)KT_IN * 4;          // tiles per weight
  const long TWv   = (long)KT_IN * 26;         // N 400 -> 416 (26 tiles)
  const long TWq   = (long)KT_IN * 4;
  const long TWi2h = (long)KT_V  * 150;
  const long TWh2h = (long)KT_IN * 150;
  const long TWck  = (long)KT_IN * 8;
  const long TWcv  = (long)KT_IN * 150;
  const long TWco  = (long)KT_C  * 38;         // N 600 -> 608 (38 tiles)

  unsigned short* Wk_p   = (unsigned short*)take((size_t)TWk   * 512);
  unsigned short* Wv_p   = (unsigned short*)take((size_t)TWv   * 512);
  unsigned short* Wq_p   = (unsigned short*)take((size_t)TWq   * 512 * NU);
  unsigned short* Wi2h_p = (unsigned short*)take((size_t)TWi2h * 512 * NU);
  unsigned short* Wh2h_p = (unsigned short*)take((size_t)TWh2h * 512 * NU);
  unsigned short* Wck_p  = (unsigned short*)take((size_t)TWck  * 512 * NU);
  unsigned short* Wcq_p  = (unsigned short*)take((size_t)TWck  * 512 * NU);
  unsigned short* Wcv_p  = (unsigned short*)take((size_t)TWcv  * 512 * NU);
  unsigned short* Wco_p  = (unsigned short*)take((size_t)TWco  * 512 * NU);

  unsigned short* x_bf    = (unsigned short*)take((size_t)BATCH * KIN_P * 2);
  unsigned short* hs_bf   = (unsigned short*)take((size_t)BATCH * NU * KIN_P * 2);
  float*          key_f   = (float*)take((size_t)BATCH * 64 * 4);
  float*          val_f   = (float*)take((size_t)BATCH * VD * 4);
  float*          qry_f   = (float*)take((size_t)BATCH * NU * 64 * 4);
  float*          mask_f  = (float*)take((size_t)BATCH * NU * 4);
  unsigned short* in_bf   = (unsigned short*)take((size_t)BATCH * NU * KV_P * 2);
  unsigned short* pre_bf  = (unsigned short*)take((size_t)BATCH * NU * H4 * 2);
  unsigned short* hb_bf   = (unsigned short*)take((size_t)BATCH * NU * KIN_P * 2);
  float*          ck_f    = (float*)take((size_t)BATCH * NU * 128 * 4);
  float*          cq_f    = (float*)take((size_t)BATCH * NU * 128 * 4);
  unsigned short* cv_bf   = (unsigned short*)take((size_t)BATCH * NU * H4 * 2);
  unsigned short* ctx_bf  = (unsigned short*)take((size_t)BATCH * NU * H4 * 2);
  if (off > ws_size) return;   // needs ~546 MB scratch

  // ---- 1) repack weights to fragment-major bf16 ----
  repack_w_k<<<dim3((unsigned)TWk, 1),    256, 0, stream>>>(Wk,   0,                Wk_p,   0,          IN, 64,   KT_IN);
  repack_w_k<<<dim3((unsigned)TWv, 1),    256, 0, stream>>>(Wv,   0,                Wv_p,   0,          IN, VD,   KT_IN);
  repack_w_k<<<dim3((unsigned)TWq, NU),   256, 0, stream>>>(Wq,   (long)HH * 64,    Wq_p,   TWq * 256,  HH, 64,   KT_IN);
  repack_w_k<<<dim3((unsigned)TWi2h, NU), 256, 0, stream>>>(Wi2h, (long)VD * H4,    Wi2h_p, TWi2h * 256, VD, H4,  KT_V);
  repack_w_k<<<dim3((unsigned)TWh2h, NU), 256, 0, stream>>>(Wh2h, (long)HH * H4,    Wh2h_p, TWh2h * 256, HH, H4,  KT_IN);
  repack_w_k<<<dim3((unsigned)TWck, NU),  256, 0, stream>>>(Wck,  (long)HH * 128,   Wck_p,  TWck * 256, HH, 128,  KT_IN);
  repack_w_k<<<dim3((unsigned)TWck, NU),  256, 0, stream>>>(Wcq,  (long)HH * 128,   Wcq_p,  TWck * 256, HH, 128,  KT_IN);
  repack_w_k<<<dim3((unsigned)TWcv, NU),  256, 0, stream>>>(Wcv,  (long)HH * H4,    Wcv_p,  TWcv * 256, HH, H4,   KT_IN);
  repack_w_k<<<dim3((unsigned)TWco, NU),  256, 0, stream>>>(Wco,  (long)H4 * HH,    Wco_p,  TWco * 256, H4, HH,   KT_C);

  // ---- 2) pack activations ----
  {
    long tx = (long)BATCH * KIN_P;
    pack_act_k<<<(unsigned)((tx + 255) / 256), 256, 0, stream>>>(x, x_bf, BATCH, IN, KIN_P);
    long th = (long)BATCH * NU * KIN_P;
    pack_act_k<<<(unsigned)((th + 255) / 256), 256, 0, stream>>>(hs, hs_bf, (long)BATCH * NU, HH, KIN_P);
  }

  const unsigned MY = BATCH / 256;   // 16 M-blocks

  // ---- 3) key = x@Wk, val = x@Wv, qry[u] = hs@Wq[u] ----
  wmma_gemm_k<<<dim3(2, MY, 1), 256, 0, stream>>>(
      x_bf, 0, KIN_P, KIN_P / 32, Wk_p, 0, KT_IN,
      nullptr, 0, 0, 0, nullptr, 0, 0,
      key_f, nullptr, 0, 64, 64);
  wmma_gemm_k<<<dim3(13, MY, 1), 256, 0, stream>>>(
      x_bf, 0, KIN_P, KIN_P / 32, Wv_p, 0, KT_IN,
      nullptr, 0, 0, 0, nullptr, 0, 0,
      val_f, nullptr, 0, VD, VD);
  wmma_gemm_k<<<dim3(2, MY, NU), 256, 0, stream>>>(
      hs_bf, KIN_P, (long)NU * KIN_P, KIN_P / 32, Wq_p, TWq * 256, KT_IN,
      nullptr, 0, 0, 0, nullptr, 0, 0,
      qry_f, nullptr, 64, (long)NU * 64, 64);

  // ---- 4) scores / top-4 mask / inputs ----
  scores_inputs_k<<<BATCH, 32, 0, stream>>>(qry_f, key_f, val_f, mask_f, in_bf);

  // ---- 5) preact = inputs@Wi2h + hs@Wh2h (fused dual-pass GEMM) ----
  wmma_gemm_k<<<dim3(75, MY, NU), 256, 0, stream>>>(
      in_bf, KV_P, (long)NU * KV_P, KV_P / 32, Wi2h_p, TWi2h * 256, KT_V,
      hs_bf, KIN_P, (long)NU * KIN_P, KIN_P / 32, Wh2h_p, TWh2h * 256, KT_IN,
      nullptr, pre_bf, H4, (long)NU * H4, H4);

  // ---- 6) LSTM gates; cs_out -> d_out; h_b -> bf16 ----
  lstm_k<<<BATCH * NU, 128, 0, stream>>>(pre_bf, cs, mask_f, hb_bf,
                                         out + (long)BATCH * NU * HH);

  // ---- 7) ck / cq / cv projections of h_b ----
  wmma_gemm_k<<<dim3(4, MY, NU), 256, 0, stream>>>(
      hb_bf, KIN_P, (long)NU * KIN_P, KIN_P / 32, Wck_p, TWck * 256, KT_IN,
      nullptr, 0, 0, 0, nullptr, 0, 0,
      ck_f, nullptr, 128, (long)NU * 128, 128);
  wmma_gemm_k<<<dim3(4, MY, NU), 256, 0, stream>>>(
      hb_bf, KIN_P, (long)NU * KIN_P, KIN_P / 32, Wcq_p, TWck * 256, KT_IN,
      nullptr, 0, 0, 0, nullptr, 0, 0,
      cq_f, nullptr, 128, (long)NU * 128, 128);
  wmma_gemm_k<<<dim3(75, MY, NU), 256, 0, stream>>>(
      hb_bf, KIN_P, (long)NU * KIN_P, KIN_P / 32, Wcv_p, TWcv * 256, KT_IN,
      nullptr, 0, 0, 0, nullptr, 0, 0,
      nullptr, cv_bf, H4, (long)NU * H4, H4);

  // ---- 8) 4-head 6x6 communication attention -> ctx ----
  cattn_k<<<BATCH * NHC, 64, 0, stream>>>(ck_f, cq_f, cv_bf, mask_f, ctx_bf);

  // ---- 9) hs_out = mask ? ctx@Wco + h_b : hs  -> d_out ----
  wmma_gemm_final_k<<<dim3(19, MY, NU), 256, 0, stream>>>(
      ctx_bf, Wco_p, TWco * 256, hb_bf, hs, mask_f, out);
}
